// ScoreModel_67345087201373
// MI455X (gfx1250) — matrix operand order repaired and checked by reading, not compiled, not run on hardware
//
#include <hip/hip_runtime.h>
#include <hip/hip_bf16.h>
#include <math.h>

#define B_  64
#define P_  1024
#define C_  128
#define H_  768
#define NH_ 12
#define HD_ 64

typedef __attribute__((ext_vector_type(16))) _Float16 v16h;
typedef __attribute__((ext_vector_type(8)))  _Float16 v8h;
typedef __attribute__((ext_vector_type(8)))  float    v8f;

// ---------------- WMMA fragment loaders (CDNA5 16x16x32 f16 layouts) ----------------
// A: 16x32 f16, row M = lane%16; half h holds K = (h<8 ? h : h+8) + (lane>=16 ? 8 : 0)
// => per lane: two contiguous 16B runs -> 2x ds_load_b128
__device__ __forceinline__ v16h frag_a(const _Float16* Ab, int lane, int ld) {
  int m = lane & 15;
  int g = (lane >> 4) << 3;
  const _Float16* r = Ab + m * ld + g;
  v8h lo = *(const v8h*)(r);
  v8h hi = *(const v8h*)(r + 16);
  return __builtin_shufflevector(lo, hi, 0, 1, 2, 3, 4, 5, 6, 7, 8, 9, 10, 11, 12, 13, 14, 15);
}
// B fragment from K-contiguous (column-major / transposed) storage:
// B(k,n) = Src[n*ld + k]; col N = lane%16; half h holds K = (h<8 ? h : h+8) + (lane>=16 ? 8 : 0)
// => per lane: two contiguous 16B runs -> 2x ds_load_b128
__device__ __forceinline__ v16h frag_bt(const _Float16* Bt, int lane, int ld) {
  int n = lane & 15;
  int g = (lane >> 4) << 3;
  const _Float16* r = Bt + (size_t)n * ld + g;
  v8h lo = *(const v8h*)(r);
  v8h hi = *(const v8h*)(r + 16);
  return __builtin_shufflevector(lo, hi, 0, 1, 2, 3, 4, 5, 6, 7, 8, 9, 10, 11, 12, 13, 14, 15);
}

// ---------------- weight convert f32 -> f16 ----------------
__global__ void cvt_kernel(const float* __restrict__ s, _Float16* __restrict__ d, int n) {
  int i = blockIdx.x * 256 + threadIdx.x;
  if (i < n) d[i] = (_Float16)s[i];
}

// ---------------- masked dot-product scores [B,P] ----------------
__global__ __launch_bounds__(256) void scores_kernel(
    const float* __restrict__ q, const float* __restrict__ corpus,
    const unsigned char* __restrict__ pm, float* __restrict__ sc) {
  int lane = threadIdx.x & 31, w = threadIdx.x >> 5;
  int pg = blockIdx.x * 8 + w;          // global (b,p)
  int b = pg >> 10, p = pg & 1023;
  float a = 0.f;
  if (pm[b * P_ + p]) {
    const float* qr = q + (size_t)b * H_;
    const float* cr = corpus + ((size_t)(b * P_ + p)) * H_;
    for (int i = lane * 4; i < H_; i += 128) {
      float4 qv = *(const float4*)&qr[i];
      float4 cv = *(const float4*)&cr[i];
      a += qv.x * cv.x + qv.y * cv.y + qv.z * cv.z + qv.w * cv.w;
    }
  }
  for (int o = 16; o > 0; o >>= 1) a += __shfl_down(a, o, 32);
  if (lane == 0) sc[pg] = a;
}

// ---------------- per-row bitonic top-128 (descending, ties -> lower index) ----------------
__global__ __launch_bounds__(512) void topk_kernel(
    const float* __restrict__ sc, const unsigned char* __restrict__ pm,
    int* __restrict__ sidx, int* __restrict__ smask) {
  __shared__ float sv[1024];
  __shared__ int   si[1024];
  int b = blockIdx.x, t = threadIdx.x;
  for (int i = t; i < 1024; i += 512) { sv[i] = sc[b * 1024 + i]; si[i] = i; }
  for (int k2 = 2; k2 <= 1024; k2 <<= 1) {
    for (int j = k2 >> 1; j > 0; j >>= 1) {
      __syncthreads();
      for (int i = t; i < 1024; i += 512) {
        int x = i ^ j;
        if (x > i) {
          float va = sv[i], vb = sv[x];
          int   ia = si[i], ib = si[x];
          bool before = (va > vb) || (va == vb && ia < ib);
          bool descBlock = ((i & k2) == 0);
          if (descBlock ? !before : before) {
            sv[i] = vb; sv[x] = va; si[i] = ib; si[x] = ia;
          }
        }
      }
    }
  }
  __syncthreads();
  for (int i = t; i < C_; i += 512) {
    int id = si[i];
    sidx[b * C_ + i]  = id;
    smask[b * C_ + i] = pm[b * 1024 + id] ? 1 : 0;
  }
}

// ---------------- build mixer input [B*C, 2H] f16: concat(q, masked cand emb) ----------------
__global__ __launch_bounds__(256) void buildx0_kernel(
    const float* __restrict__ q, const float* __restrict__ corpus,
    const int* __restrict__ sidx, const int* __restrict__ smask,
    _Float16* __restrict__ X0) {
  int bc = blockIdx.x, t = threadIdx.x;
  int b = bc >> 7;
  int id = sidx[bc];
  float mk = smask[bc] ? 1.f : 0.f;
  _Float16* xr = X0 + (size_t)bc * (2 * H_);
  const float* qr = q + (size_t)b * H_;
  const float* cr = corpus + ((size_t)(b * P_ + id)) * H_;
  for (int i = t; i < H_; i += 256) {
    xr[i]      = (_Float16)qr[i];
    xr[H_ + i] = (_Float16)(cr[i] * mk);
  }
}

// LDS pitches (halves): pitch*2 bytes stays a multiple of 16 for b128 alignment
#define LDA_ 40   // 64x32 A tile, padded
#define LDB_ 40   // 64(n) x 32(k) transposed B tile, padded

// ---------------- WMMA GEMM core macro body (A staged via async DMA, B transposed) -------
#define GEMM_CORE()                                                                         \
  __shared__ __align__(16) _Float16 As[64 * LDA_];                                          \
  __shared__ __align__(16) _Float16 Bt[64 * LDB_];                                          \
  int t = threadIdx.x, lane = t & 31, wave = t >> 5;                                        \
  int wm = wave & 3, wn = wave >> 2;                                                        \
  int m0 = blockIdx.y * 64, n0 = blockIdx.x * 64;                                           \
  v8f acc0 = {}, acc1 = {};                                                                 \
  int arow = t >> 2, acol = (t & 3) * 8;                                                    \
  int brow = t >> 3, bcol = (t & 7) * 8;                                                    \
  unsigned lA = (unsigned)(size_t)&As[arow * LDA_ + acol];                                  \
  for (int kk = 0; kk < K; kk += 32) {                                                      \
    __syncthreads();                                                                        \
    unsigned long long ga =                                                                 \
        (unsigned long long)(size_t)&A[(size_t)(m0 + arow) * K + kk + acol];                \
    asm volatile("global_load_async_to_lds_b128 %0, %1, off" ::"v"(lA), "v"(ga)             \
                 : "memory");                                                               \
    v8h wv = *(const v8h*)&W[(size_t)(kk + brow) * N + n0 + bcol];                          \
    _Pragma("unroll") for (int j = 0; j < 8; ++j) Bt[(bcol + j) * LDB_ + brow] = wv[j];     \
    if (kk + 32 < K) {                                                                      \
      __builtin_prefetch(&A[(size_t)(m0 + arow) * K + kk + 32 + acol], 0, 1);               \
      __builtin_prefetch(&W[(size_t)(kk + 32 + brow) * N + n0 + bcol], 0, 1);               \
    }                                                                                       \
    asm volatile("s_wait_asynccnt 0" ::: "memory");                                         \
    __syncthreads();                                                                        \
    v16h a  = frag_a(&As[wm * 16 * LDA_], lane, LDA_);                                      \
    v16h b0 = frag_bt(&Bt[(wn * 32) * LDB_], lane, LDB_);                                   \
    v16h b1 = frag_bt(&Bt[(wn * 32 + 16) * LDB_], lane, LDB_);                              \
    acc0 = __builtin_amdgcn_wmma_f32_16x16x32_f16(false, a, false, b0, (short)0, acc0,      \
                                                  false, false);                            \
    acc1 = __builtin_amdgcn_wmma_f32_16x16x32_f16(false, a, false, b1, (short)0, acc1,      \
                                                  false, false);                            \
  }                                                                                         \
  int mloc = m0 + wm * 16 + ((lane >> 4) << 3);                                             \
  int nl = lane & 15;

// ---------------- WMMA GEMM, f16 out (act: 0 none, 1 relu, 2 tanh) ----------------
__global__ __launch_bounds__(256) void gemm_f16out(
    const _Float16* __restrict__ A, const _Float16* __restrict__ W,
    const float* __restrict__ bias, _Float16* __restrict__ Y,
    int M, int N, int K, int act) {
  GEMM_CORE()
#pragma unroll
  for (int r = 0; r < 8; ++r) {
    int m = mloc + r;
    {
      int n = n0 + wn * 32 + nl;
      float v = acc0[r] + bias[n];
      if (act == 1) v = fmaxf(v, 0.f);
      else if (act == 2) v = tanhf(v);
      Y[(size_t)m * N + n] = (_Float16)v;
    }
    {
      int n = n0 + wn * 32 + 16 + nl;
      float v = acc1[r] + bias[n];
      if (act == 1) v = fmaxf(v, 0.f);
      else if (act == 2) v = tanhf(v);
      Y[(size_t)m * N + n] = (_Float16)v;
    }
  }
}

// ---------------- WMMA GEMM -> f32 out, optional ReLU and residual add ----------------
__global__ __launch_bounds__(256) void gemm_f32out(
    const _Float16* __restrict__ A, const _Float16* __restrict__ W,
    const float* __restrict__ bias, const float* __restrict__ res,
    float* __restrict__ Y, int M, int N, int K, int act) {
  GEMM_CORE()
#pragma unroll
  for (int r = 0; r < 8; ++r) {
    int m = mloc + r;
    {
      int n = n0 + wn * 32 + nl;
      float v = acc0[r] + bias[n];
      if (act == 1) v = fmaxf(v, 0.f);
      if (res) v += res[(size_t)m * N + n];
      Y[(size_t)m * N + n] = v;
    }
    {
      int n = n0 + wn * 32 + 16 + nl;
      float v = acc1[r] + bias[n];
      if (act == 1) v = fmaxf(v, 0.f);
      if (res) v += res[(size_t)m * N + n];
      Y[(size_t)m * N + n] = v;
    }
  }
}

// ---------------- LayerNorm (fp32) -> f32 out + f16 copy ----------------
__global__ __launch_bounds__(256) void ln_kernel(
    const float* __restrict__ X, const float* __restrict__ g,
    const float* __restrict__ bb, float* __restrict__ Yf,
    _Float16* __restrict__ Yh) {
  __shared__ float red[256];
  int row = blockIdx.x, t = threadIdx.x;
  const float* x = X + (size_t)row * H_;
  float s = 0.f;
  for (int i = t; i < H_; i += 256) s += x[i];
  red[t] = s; __syncthreads();
  for (int o = 128; o > 0; o >>= 1) { if (t < o) red[t] += red[t + o]; __syncthreads(); }
  float mu = red[0] * (1.f / H_);
  __syncthreads();
  float v = 0.f;
  for (int i = t; i < H_; i += 256) { float d = x[i] - mu; v += d * d; }
  red[t] = v; __syncthreads();
  for (int o = 128; o > 0; o >>= 1) { if (t < o) red[t] += red[t + o]; __syncthreads(); }
  float rs = rsqrtf(red[0] * (1.f / H_) + 1e-5f);
  for (int i = t; i < H_; i += 256) {
    float y = (x[i] - mu) * rs * g[i] + bb[i];
    Yf[(size_t)row * H_ + i] = y;
    Yh[(size_t)row * H_ + i] = (_Float16)y;
  }
}

// attention LDS pitches (halves)
#define LQK_ 72    // q,k tiles [128][72]
#define LVP_ 136   // v^T [64][136] and P [128][136]

// ---------------- attention per (b, head): WMMA qk^T, masked softmax, WMMA Pv ----------------
__global__ __launch_bounds__(256) void attn_kernel(
    const _Float16* __restrict__ qkv, const int* __restrict__ cmask,
    _Float16* __restrict__ ctx) {
  extern __shared__ char smem[];
  _Float16* qs  = (_Float16*)smem;                 // [128][LQK_]
  _Float16* ks  = qs + 128 * LQK_;                 // [128][LQK_]
  _Float16* vst = ks + 128 * LQK_;                 // [64][LVP_]   v transposed: [hd][cand]
  float*    Sf  = (float*)(vst + 64 * LVP_);       // [128][128]
  _Float16* Ph  = (_Float16*)(Sf + 128 * 128);     // [128][LVP_]
  int b = blockIdx.y, hd = blockIdx.x;
  int t = threadIdx.x, lane = t & 31, w = t >> 5;
  const _Float16* base = qkv + (size_t)b * C_ * (3 * H_);
  // stage q,k via async DMA; v transposed via regular load + scatter
  for (int i = t; i < 1024; i += 256) {
    int row = i >> 3, c8 = (i & 7) * 8;
    size_t roff = (size_t)row * (3 * H_) + hd * HD_ + c8;
    unsigned lq = (unsigned)(size_t)&qs[row * LQK_ + c8];
    unsigned lk = (unsigned)(size_t)&ks[row * LQK_ + c8];
    unsigned long long gq = (unsigned long long)(size_t)&base[roff];
    unsigned long long gk = (unsigned long long)(size_t)&base[roff + H_];
    asm volatile("global_load_async_to_lds_b128 %0, %1, off" ::"v"(lq), "v"(gq) : "memory");
    asm volatile("global_load_async_to_lds_b128 %0, %1, off" ::"v"(lk), "v"(gk) : "memory");
    v8h vv = *(const v8h*)&base[roff + 2 * H_];
#pragma unroll
    for (int j = 0; j < 8; ++j) vst[(c8 + j) * LVP_ + row] = vv[j];
  }
  asm volatile("s_wait_asynccnt 0" ::: "memory");
  __syncthreads();
  // S = q k^T / sqrt(64), masked
  v8f acc[8] = {};
  for (int kk = 0; kk < 64; kk += 32) {
    v16h a = frag_a(&qs[w * 16 * LQK_ + kk], lane, LQK_);
#pragma unroll
    for (int j = 0; j < 8; ++j) {
      v16h bb = frag_bt(&ks[(j * 16) * LQK_ + kk], lane, LQK_);
      acc[j] = __builtin_amdgcn_wmma_f32_16x16x32_f16(false, a, false, bb, (short)0, acc[j], false, false);
    }
  }
  int hi = (lane >> 4) << 3, nl = lane & 15;
  const int* cm = cmask + b * C_;
#pragma unroll
  for (int j = 0; j < 8; ++j) {
    int n = j * 16 + nl;
    int mk = cm[n];
#pragma unroll
    for (int r = 0; r < 8; ++r) {
      float v = acc[j][r] * 0.125f;
      if (!mk) v = -1e9f;
      Sf[(w * 16 + hi + r) * 128 + n] = v;
    }
  }
  __syncthreads();
  // row softmax (fp32)
  if (t < 128) {
    float mx = -3.4e38f;
    for (int n = 0; n < 128; ++n) mx = fmaxf(mx, Sf[t * 128 + n]);
    float s = 0.f;
    for (int n = 0; n < 128; ++n) { float e = __expf(Sf[t * 128 + n] - mx); Sf[t * 128 + n] = e; s += e; }
    float inv = 1.f / s;
    for (int n = 0; n < 128; ++n) Ph[t * LVP_ + n] = (_Float16)(Sf[t * 128 + n] * inv);
  }
  __syncthreads();
  // O = P @ v   (B fragment from v^T, K-contiguous)
  v8f oc[4] = {};
  for (int kk = 0; kk < 128; kk += 32) {
    v16h a = frag_a(&Ph[w * 16 * LVP_ + kk], lane, LVP_);
#pragma unroll
    for (int j = 0; j < 4; ++j) {
      v16h bb = frag_bt(&vst[(j * 16) * LVP_ + kk], lane, LVP_);
      oc[j] = __builtin_amdgcn_wmma_f32_16x16x32_f16(false, a, false, bb, (short)0, oc[j], false, false);
    }
  }
#pragma unroll
  for (int j = 0; j < 4; ++j) {
    int n = j * 16 + nl;
#pragma unroll
    for (int r = 0; r < 8; ++r) {
      int m = w * 16 + hi + r;
      ctx[((size_t)(b * C_ + m)) * H_ + hd * HD_ + n] = (_Float16)oc[j][r];
    }
  }
}

// ---------------- decoder head + outputs ----------------
__global__ __launch_bounds__(256) void dec2_kernel(
    const _Float16* __restrict__ Hd, const float* __restrict__ w2,
    const float* __restrict__ b2, const int* __restrict__ smask,
    const int* __restrict__ sidx, float* __restrict__ out) {
  __shared__ float red[256];
  int bc = blockIdx.x, t = threadIdx.x;
  float s = 0.f;
  for (int i = t; i < H_; i += 256) s += (float)Hd[(size_t)bc * H_ + i] * w2[i];
  red[t] = s; __syncthreads();
  for (int o = 128; o > 0; o >>= 1) { if (t < o) red[t] += red[t + o]; __syncthreads(); }
  if (t == 0) {
    float z = red[0] + b2[0];
    float lik = 1.f / (1.f + __expf(-z));
    int mk = smask[bc];
    out[bc] = mk ? lik : 0.f;
    out[B_ * C_ + bc] = (float)mk;
    out[2 * B_ * C_ + bc] = (float)sidx[bc];
  }
}

extern "C" void kernel_launch(void* const* d_in, const int* in_sizes, int n_in,
                              void* d_out, int out_size, void* d_ws, size_t ws_size,
                              hipStream_t stream) {
  (void)in_sizes; (void)n_in; (void)out_size; (void)ws_size;
  const float* q          = (const float*)d_in[0];
  const float* corpus     = (const float*)d_in[1];
  const unsigned char* pm = (const unsigned char*)d_in[2];
  const float* mixer_w    = (const float*)d_in[3];
  const float* mixer_b    = (const float*)d_in[4];
  const float* mixer_ln_g = (const float*)d_in[5];
  const float* mixer_ln_b = (const float*)d_in[6];
  const float* attn_in_w  = (const float*)d_in[7];
  const float* attn_in_b  = (const float*)d_in[8];
  const float* attn_out_w = (const float*)d_in[9];
  const float* attn_out_b = (const float*)d_in[10];
  const float* attn_ln_g  = (const float*)d_in[11];
  const float* attn_ln_b  = (const float*)d_in[12];
  const float* ffn_w1     = (const float*)d_in[13];
  const float* ffn_b1     = (const float*)d_in[14];
  const float* ffn_w2     = (const float*)d_in[15];
  const float* ffn_b2     = (const float*)d_in[16];
  const float* ffn_ln_g   = (const float*)d_in[17];
  const float* ffn_ln_b   = (const float*)d_in[18];
  const float* dec_w1     = (const float*)d_in[19];
  const float* dec_b1     = (const float*)d_in[20];
  const float* dec_w2     = (const float*)d_in[21];
  const float* dec_b2     = (const float*)d_in[22];

  char* ws = (char*)d_ws;
  size_t off = 0;
  auto alloc = [&](size_t bytes) -> void* {
    void* p = ws + off;
    off = (off + bytes + 255) & ~(size_t)255;
    return p;
  };
  _Float16* w_mix16 = (_Float16*)alloc((size_t)1536 * 768 * 2);
  _Float16* w_qkv16 = (_Float16*)alloc((size_t)768 * 2304 * 2);
  _Float16* w_ao16  = (_Float16*)alloc((size_t)768 * 768 * 2);
  _Float16* w_f116  = (_Float16*)alloc((size_t)768 * 3072 * 2);
  _Float16* w_f216  = (_Float16*)alloc((size_t)3072 * 768 * 2);
  _Float16* w_d116  = (_Float16*)alloc((size_t)768 * 768 * 2);
  float*    scores  = (float*)alloc((size_t)B_ * P_ * 4);
  int*      sidx    = (int*)alloc((size_t)B_ * C_ * 4);
  int*      smask   = (int*)alloc((size_t)B_ * C_ * 4);
  float*    xa      = (float*)alloc((size_t)B_ * C_ * H_ * 4);
  float*    xb      = (float*)alloc((size_t)B_ * C_ * H_ * 4);
  _Float16* xh      = (_Float16*)alloc((size_t)B_ * C_ * H_ * 2);
  _Float16* bufA    = (_Float16*)alloc((size_t)B_ * C_ * 3072 * 2);
  _Float16* bufB    = (_Float16*)alloc((size_t)B_ * C_ * 2304 * 2);

  auto cvt = [&](const float* s, _Float16* d, int n) {
    cvt_kernel<<<(n + 255) / 256, 256, 0, stream>>>(s, d, n);
  };
  cvt(mixer_w, w_mix16, 1536 * 768);
  cvt(attn_in_w, w_qkv16, 768 * 2304);
  cvt(attn_out_w, w_ao16, 768 * 768);
  cvt(ffn_w1, w_f116, 768 * 3072);
  cvt(ffn_w2, w_f216, 3072 * 768);
  cvt(dec_w1, w_d116, 768 * 768);

  const int M = B_ * C_;  // 8192
  scores_kernel<<<B_ * P_ / 8, 256, 0, stream>>>(q, corpus, pm, scores);
  topk_kernel<<<B_, 512, 0, stream>>>(scores, pm, sidx, smask);
  buildx0_kernel<<<M, 256, 0, stream>>>(q, corpus, sidx, smask, bufB);

  // mixer: relu(X0 @ Wmix + b) -> xa (f32), then LN -> xb(f32)+xh(f16)
  gemm_f32out<<<dim3(768 / 64, M / 64), 256, 0, stream>>>(bufB, w_mix16, mixer_b, nullptr, xa, M, 768, 1536, 1);
  ln_kernel<<<M, 256, 0, stream>>>(xa, mixer_ln_g, mixer_ln_b, xb, xh);
  // qkv
  gemm_f16out<<<dim3(2304 / 64, M / 64), 256, 0, stream>>>(xh, w_qkv16, attn_in_b, bufA, M, 2304, 768, 0);
  // attention -> ctx (bufB)
  size_t attn_lds = (size_t)(128 * LQK_ * 2) * 2 + (size_t)64 * LVP_ * 2 + (size_t)128 * 128 * 4 + (size_t)128 * LVP_ * 2;
  attn_kernel<<<dim3(NH_, B_), 256, attn_lds, stream>>>(bufA, smask, bufB);
  // out-proj + residual(mixer_out=xb) -> xa, LN -> xb+xh
  gemm_f32out<<<dim3(768 / 64, M / 64), 256, 0, stream>>>(bufB, w_ao16, attn_out_b, xb, xa, M, 768, 768, 0);
  ln_kernel<<<M, 256, 0, stream>>>(xa, attn_ln_g, attn_ln_b, xb, xh);
  // FFN
  gemm_f16out<<<dim3(3072 / 64, M / 64), 256, 0, stream>>>(xh, w_f116, ffn_b1, bufA, M, 3072, 768, 1);
  gemm_f32out<<<dim3(768 / 64, M / 64), 256, 0, stream>>>(bufA, w_f216, ffn_b2, xb, xa, M, 768, 3072, 0);
  ln_kernel<<<M, 256, 0, stream>>>(xa, ffn_ln_g, ffn_ln_b, xb, xh);
  // decoder
  gemm_f16out<<<dim3(768 / 64, M / 64), 256, 0, stream>>>(xh, w_d116, dec_b1, bufB, M, 768, 768, 2);
  dec2_kernel<<<M, 256, 0, stream>>>(bufB, dec_w2, dec_b2, smask, sidx, (float*)d_out);
}
